// CachedLightningIndexer_57604101374287
// MI455X (gfx1250) — matrix operand order repaired and checked by reading, not compiled, not run on hardware
//
#include <hip/hip_runtime.h>

// ---------------------------------------------------------------------------
// CachedLightningIndexer for gfx1250 (MI455X, wave32, WMMA)
//
// out[b,t,s] = sum_h relu( (x@wq)[b,t,h,:] . (x@wk)[b,s,:] ) * (x@ww)[b,t,h]
//
// Phase 1: cast x -> f16, pack/transpose weights -> f16
// Phase 2: fused projection GEMM via v_wmma_f32_16x16x32_f16 (K=2048)
// Phase 3: dots + relu + head-combine via v_wmma_f32_16x16x32_f16 (K=d=32,
//          exactly one WMMA per head per 16x16 tile). Store-bound: 128 MiB
//          output / 23.3 TB/s ~ 5.5 us floor; everything else L2-resident.
// ---------------------------------------------------------------------------

typedef _Float16 v16h __attribute__((ext_vector_type(16)));
typedef _Float16 v8h  __attribute__((ext_vector_type(8)));
typedef _Float16 v4h  __attribute__((ext_vector_type(4)));
typedef float    v8f  __attribute__((ext_vector_type(8)));

#define NB   2
#define NT   4096          // t (== s)
#define DM   2048          // d_model
#define NH   2             // heads
#define DH   32            // head dim
#define NCOL 112           // 64 (Q) + 32 (K) + 2 (W) padded to 7*16

static __device__ __forceinline__ v16h cat16(v8h lo, v8h hi) {
  return __builtin_shufflevector(lo, hi, 0,1,2,3,4,5,6,7,8,9,10,11,12,13,14,15);
}

static __device__ __forceinline__ v8f wmma_f16(v16h a, v16h b, v8f c) {
  // D(16x16 f32) = A(16x32 f16) x B(32x16 f16) + C
  return __builtin_amdgcn_wmma_f32_16x16x32_f16(false, a, false, b,
                                                (short)0, c, false, false);
}

// ---------------------------------------------------------------------------
// Kernel 1: x (f32) -> xh (f16), same [b*t][d_model] row-major layout
// ---------------------------------------------------------------------------
__global__ void k_cvt_x(const float* __restrict__ x, _Float16* __restrict__ xh,
                        int n4) {
  int i = blockIdx.x * blockDim.x + threadIdx.x;
  if (i >= n4) return;
  float4 v = ((const float4*)x)[i];
  v4h o = { (_Float16)v.x, (_Float16)v.y, (_Float16)v.z, (_Float16)v.w };
  ((v4h*)xh)[i] = o;
}

// ---------------------------------------------------------------------------
// Kernel 2: build transposed packed weight matrix wt[n][k] (f16), n in [0,112)
//   n <  64 : wq column n           (wq is [2048][64] row-major)
//   n <  96 : wk column n-64        (wk is [2048][32])
//   n <  98 : ww column n-96        (ww is [2048][2])
//   else    : zero padding (computed in N-tile 6, dropped by the k_proj epilogue)
// ---------------------------------------------------------------------------
__global__ void k_build_w(const float* __restrict__ wq,
                          const float* __restrict__ wk,
                          const float* __restrict__ ww,
                          _Float16* __restrict__ wt) {
  int i = blockIdx.x * blockDim.x + threadIdx.x;
  if (i >= NCOL * DM) return;
  int n = i >> 11;          // / 2048
  int k = i & (DM - 1);
  float v = 0.f;
  if (n < 64)       v = wq[k * 64 + n];
  else if (n < 96)  v = wk[k * 32 + (n - 64)];
  else if (n < 98)  v = ww[k * 2  + (n - 96)];
  wt[(size_t)n * DM + k] = (_Float16)v;
}

// ---------------------------------------------------------------------------
// Kernel 3: fused projection GEMM  P[8192 x 112] = xh[8192 x 2048] @ Wall
// One wave per 16x16 output tile; 64 WMMA k-steps. Epilogue scatters the
// D-matrix elements (lane holds m = r + 8*(lane>>4), n = lane%16) into
//   qh : f16 [b][h][t][32]   (A-friendly layout for phase 4)
//   kh : f16 [b][s][32]
//   wf : f32 [b][t][h]       (kept f32 — it scales the relu'd dots)
// ---------------------------------------------------------------------------
__global__ void k_proj(const _Float16* __restrict__ xh,
                       const _Float16* __restrict__ wt,
                       _Float16* __restrict__ qh,
                       _Float16* __restrict__ kh,
                       float*    __restrict__ wf) {
  int gtid = blockIdx.x * blockDim.x + threadIdx.x;
  int w    = gtid >> 5;
  int lane = threadIdx.x & 31;
  int lm   = lane & 15;
  int hi   = lane >> 4;

  int mt = w / 7;               // 512 M-tiles (rows of x)
  int nt = w - mt * 7;          // 7 N-tiles (columns 0..111)

  const _Float16* arow = xh + (size_t)(mt * 16 + lm) * DM;
  const _Float16* brow = wt + (size_t)(nt * 16 + lm) * DM;

  v8f acc = {0.f, 0.f, 0.f, 0.f, 0.f, 0.f, 0.f, 0.f};

  for (int k = 0; k < DM; k += 32) {
    // A 16x32 f16: lanes 0-15 take K chunks {0..7},{16..23}; lanes 16-31 +8
    v8h alo = *(const v8h*)(arow + k +  8 * hi);
    v8h ahi = *(const v8h*)(arow + k + 16 + 8 * hi);
    // B 32x16 f16: lanes 0-15 take K chunks {0..7},{16..23}; lanes 16-31 +8
    v8h blo = *(const v8h*)(brow + k +  8 * hi);
    v8h bhi = *(const v8h*)(brow + k + 16 + 8 * hi);
    acc = wmma_f16(cat16(alo, ahi), cat16(blo, bhi), acc);
  }

  int col = nt * 16 + lm;
  #pragma unroll
  for (int r = 0; r < 8; ++r) {
    int row = mt * 16 + r + 8 * hi;       // global token row in [0, 8192)
    int b   = row >> 12;
    int t   = row & (NT - 1);
    float v = acc[r];
    if (col < 64) {
      int h = col >> 5, d = col & 31;
      qh[((size_t)(b * NH + h) * NT + t) * DH + d] = (_Float16)v;
    } else if (col < 96) {
      kh[((size_t)b * NT + t) * DH + (col - 64)] = (_Float16)v;
    } else if (col < 98) {
      wf[((size_t)b * NT + t) * NH + (col - 96)] = v;
    }
  }
}

// ---------------------------------------------------------------------------
// Kernel 4: dots + relu + head-combine.
// One wave per 16(t) x 64(s) strip. A-tiles (Q, both heads) live in registers
// for the whole strip; per 16x16 subtile: 1 B load (K-tile), 2 WMMAs
// (K-dim = head dim = 32 -> a single v_wmma_f32_16x16x32_f16 per head),
// then out = relu(acc0)*w0 + relu(acc1)*w1, stored f32.
// ---------------------------------------------------------------------------
__global__ void k_dots(const _Float16* __restrict__ qh,
                       const _Float16* __restrict__ kh,
                       const float*    __restrict__ wf,
                       float*          __restrict__ out) {
  int gtid = blockIdx.x * blockDim.x + threadIdx.x;
  int w    = gtid >> 5;
  int lane = threadIdx.x & 31;
  int lm   = lane & 15;
  int hi   = lane >> 4;

  int b  = w >> 14;                 // 16384 strips per batch
  int r0 = w & 16383;
  int t0 = (r0 >> 6) << 4;          // 256 t-tiles of 16
  int s0 = (r0 & 63) << 6;          // 64 s-strips of 64

  // A tiles (Q) for both heads, held across the strip
  const _Float16* a0p = qh + ((size_t)((b * NH + 0) * NT + t0 + lm)) * DH + 8 * hi;
  const _Float16* a1p = qh + ((size_t)((b * NH + 1) * NT + t0 + lm)) * DH + 8 * hi;
  v16h a0 = cat16(*(const v8h*)a0p, *(const v8h*)(a0p + 16));
  v16h a1 = cat16(*(const v8h*)a1p, *(const v8h*)(a1p + 16));

  // Per-row head weights: lane's D rows are m = r + 8*hi
  float w0[8], w1[8];
  #pragma unroll
  for (int r = 0; r < 8; ++r) {
    size_t t = (size_t)b * NT + (t0 + r + 8 * hi);
    w0[r] = wf[t * NH + 0];
    w1[r] = wf[t * NH + 1];
  }

  const v8f zero = {0.f, 0.f, 0.f, 0.f, 0.f, 0.f, 0.f, 0.f};

  #pragma unroll
  for (int j = 0; j < 4; ++j) {
    int s = s0 + j * 16;
    const _Float16* bp = kh + ((size_t)(b * NT + s + lm)) * DH + 8 * hi;
    v16h bm = cat16(*(const v8h*)bp, *(const v8h*)(bp + 16));

    v8f acc0 = wmma_f16(a0, bm, zero);
    v8f acc1 = wmma_f16(a1, bm, zero);

    #pragma unroll
    for (int r = 0; r < 8; ++r) {
      float o = fmaxf(acc0[r], 0.f) * w0[r] + fmaxf(acc1[r], 0.f) * w1[r];
      out[((size_t)(b * NT + t0 + r + 8 * hi)) * NT + s + lm] = o;
    }
  }
}

// ---------------------------------------------------------------------------
// Launch
// ---------------------------------------------------------------------------
extern "C" void kernel_launch(void* const* d_in, const int* in_sizes, int n_in,
                              void* d_out, int out_size, void* d_ws, size_t ws_size,
                              hipStream_t stream) {
  const float* x  = (const float*)d_in[0];
  const float* wq = (const float*)d_in[1];
  const float* wk = (const float*)d_in[2];
  const float* ww = (const float*)d_in[3];
  float* out = (float*)d_out;

  // Workspace layout (bytes), total ~36.7 MiB
  char* ws = (char*)d_ws;
  _Float16* xh = (_Float16*)(ws);                     // 8192*2048*2   = 33,554,432
  _Float16* wt = (_Float16*)(ws + 33554432);          // 112*2048*2    =    458,752
  _Float16* qh = (_Float16*)(ws + 34013184);          // 2*2*4096*32*2 =  2,097,152
  _Float16* kh = (_Float16*)(ws + 36110336);          // 2*4096*32*2   =    524,288
  float*    wf = (float*)   (ws + 36634624);          // 2*4096*2*4    =     65,536

  // 1) cast x -> f16 (float4 vectorized)
  int n4 = NB * NT * DM / 4;                          // 4,194,304
  k_cvt_x<<<n4 / 256, 256, 0, stream>>>(x, xh, n4);

  // 2) pack + transpose weights
  k_build_w<<<(NCOL * DM) / 256, 256, 0, stream>>>(wq, wk, ww, wt);

  // 3) projection GEMM: 512 M-tiles * 7 N-tiles = 3584 waves, 8 waves/block
  k_proj<<<448, 256, 0, stream>>>(xh, wt, qh, kh, wf);

  // 4) dots: 2 * 256 * 64 = 32768 waves, 8 waves/block
  k_dots<<<4096, 256, 0, stream>>>(qh, kh, wf, out);
}
